// GCN_19756849561729
// MI455X (gfx1250) — compile-verified, matrix operand order checked by reading, not run on hardware
//
#include <hip/hip_runtime.h>
#include <hip/hip_bf16.h>

// ---------------------------------------------------------------------------
// GCN forward on MI455X (gfx1250, wave32).
//   GEMM1 (30000x1433x512) and GEMM2 (30000x512x128) run through
//   v_wmma_f32_16x16x32_bf16 with f32 accumulation. Tiles are staged to LDS
//   with a register-based software pipeline: global loads for tile k+1 are
//   issued (clamped, branch-free) while WMMAs consume tile k, so HBM/L2
//   latency hides behind the matrix pipe. SpMM uses f32 global atomics that
//   resolve in the 192MB L2 (outputs are 61MB / 15MB).
// ---------------------------------------------------------------------------

typedef __attribute__((ext_vector_type(16))) __bf16 v16bf;
typedef __attribute__((ext_vector_type(8)))  __bf16 v8bf;
typedef __attribute__((ext_vector_type(2)))  __bf16 v2bf;
typedef __attribute__((ext_vector_type(8)))  float  v8f;

#define BM 128
#define BN 128
#define BK 32
#define LDS_STRIDE 40   // BK + 8 bf16 padding to spread banks

// C = A[MxK] @ B[KxN], f32 row-major, bf16 WMMA with f32 accumulate.
// Requires N % 128 == 0 (true for all call sites: 512, 128).
// Block: 256 threads = 8 waves; wave (wm,wn) owns a 32x64 output sub-tile.
__global__ __launch_bounds__(256)
void gemm_bf16_wmma(const float* __restrict__ A, const float* __restrict__ B,
                    float* __restrict__ C, int M, int N, int K) {
  __shared__ __align__(16) __bf16 sA[BM * LDS_STRIDE];   // sA[m][k]
  __shared__ __align__(16) __bf16 sB[BN * LDS_STRIDE];   // sB[n][k] (transposed)

  const int tid  = threadIdx.x;
  const int lane = tid & 31;
  const int wave = tid >> 5;
  const int wm   = wave & 3;       // 0..3 -> 32-row strip
  const int wn   = wave >> 2;      // 0..1 -> 64-col strip
  const int m0   = blockIdx.x * BM;
  const int n0   = blockIdx.y * BN;

  const int hb  = lane >> 4;       // half-wave select
  const int l16 = lane & 15;

  v8f acc[2][4];
#pragma unroll
  for (int i = 0; i < 2; ++i)
#pragma unroll
    for (int j = 0; j < 4; ++j)
#pragma unroll
      for (int r = 0; r < 8; ++r) acc[i][j][r] = 0.0f;

  // Register staging buffers: A as 8 k-pairs/thread, B as 16 scalars/thread.
  float va0[8], va1[8], vb[16];

  // Branch-free global -> register tile fetch (addresses clamped in-range;
  // out-of-range elements are zeroed at the LDS store). Loads issue
  // back-to-back with a single batched wait at first use.
  auto loadTile = [&](int k0) {
#pragma unroll
    for (int t = 0; t < 8; ++t) {
      int idx = tid + t * 256;            // pair index 0..2047
      int m = idx >> 4, kp = idx & 15;
      int gm = m0 + m, gk = k0 + 2 * kp;
      int cm  = gm     < M ? gm     : M - 1;
      int ck0 = gk     < K ? gk     : K - 1;
      int ck1 = gk + 1 < K ? gk + 1 : K - 1;
      const float* row = A + (size_t)cm * K;
      va0[t] = row[ck0];
      va1[t] = row[ck1];
    }
#pragma unroll
    for (int t = 0; t < 16; ++t) {
      int idx = tid + t * 256;            // 0..4095
      int k = idx >> 7, n = idx & 127;    // adjacent lanes -> adjacent n
      int gk = k0 + k;
      int ck = gk < K ? gk : K - 1;
      vb[t] = B[(size_t)ck * N + (n0 + n)];
    }
  };

  auto storeTile = [&](int k0) {
#pragma unroll
    for (int t = 0; t < 8; ++t) {
      int idx = tid + t * 256;
      int m = idx >> 4, kp = idx & 15;
      int gm = m0 + m, gk = k0 + 2 * kp;
      float x0 = (gm < M && gk     < K) ? va0[t] : 0.0f;
      float x1 = (gm < M && gk + 1 < K) ? va1[t] : 0.0f;
      v2bf p; p[0] = (__bf16)x0; p[1] = (__bf16)x1;     // v_cvt_pk_bf16_f32
      *(v2bf*)&sA[m * LDS_STRIDE + 2 * kp] = p;         // ds_store_b32
    }
#pragma unroll
    for (int t = 0; t < 16; ++t) {
      int idx = tid + t * 256;
      int k = idx >> 7, n = idx & 127;
      int gk = k0 + k;
      sB[n * LDS_STRIDE + k] = (gk < K) ? (__bf16)vb[t] : (__bf16)0.0f;
    }
  };

  const int nk = (K + BK - 1) / BK;
  loadTile(0);

  for (int kt = 0; kt < nk; ++kt) {
    storeTile(kt * BK);
    __syncthreads();
    if (kt + 1 < nk) loadTile((kt + 1) * BK);   // overlap with WMMA below

    // ---- fragments per ISA 16-bit operand layout:
    // lanes 0-15: elems 0-7 = K 0-7,  elems 8-15 = K 16-23
    // lanes16-31: elems 0-7 = K 8-15, elems 8-15 = K 24-31
    union Frag { v16bf v; v8bf h[2]; };
    Frag a[2], b[4];
#pragma unroll
    for (int i = 0; i < 2; ++i) {
      const __bf16* base = &sA[(wm * 32 + i * 16 + l16) * LDS_STRIDE + hb * 8];
      a[i].h[0] = *(const v8bf*)(base);
      a[i].h[1] = *(const v8bf*)(base + 16);
    }
#pragma unroll
    for (int j = 0; j < 4; ++j) {
      const __bf16* base = &sB[(wn * 64 + j * 16 + l16) * LDS_STRIDE + hb * 8];
      b[j].h[0] = *(const v8bf*)(base);
      b[j].h[1] = *(const v8bf*)(base + 16);
    }
#pragma unroll
    for (int i = 0; i < 2; ++i)
#pragma unroll
      for (int j = 0; j < 4; ++j)
        acc[i][j] = __builtin_amdgcn_wmma_f32_16x16x32_bf16(
            false, a[i].v, false, b[j].v, (short)0, acc[i][j], false, false);
    __syncthreads();
  }

  // ---- store: C/D layout n = lane&15, m = r + 8*(lane>=16) ----
#pragma unroll
  for (int i = 0; i < 2; ++i) {
#pragma unroll
    for (int j = 0; j < 4; ++j) {
      int gn = n0 + wn * 64 + j * 16 + l16;
#pragma unroll
      for (int r = 0; r < 8; ++r) {
        int gm = m0 + wm * 32 + i * 16 + hb * 8 + r;
        if (gm < M) C[(size_t)gm * N + gn] = acc[i][j][r];
      }
    }
  }
}

// out[dst] += w * h[src] ; one thread per (edge, 4-float feature group).
__global__ __launch_bounds__(256)
void spmm_atomic(const int* __restrict__ src, const int* __restrict__ dst,
                 const float* __restrict__ w, const float* __restrict__ h,
                 float* __restrict__ out, int E, int gshift, int D) {
  long long gid = (long long)blockIdx.x * blockDim.x + threadIdx.x;
  long long total = (long long)E << gshift;
  if (gid >= total) return;
  int e = (int)(gid >> gshift);
  int g = (int)(gid & ((1 << gshift) - 1));
  int s = src[e], d = dst[e];
  float wt = w[e];
  const float4 hv = *(const float4*)(h + (size_t)s * D + g * 4);
  float* o = out + (size_t)d * D + g * 4;
  atomicAdd(o + 0, wt * hv.x);
  atomicAdd(o + 1, wt * hv.y);
  atomicAdd(o + 2, wt * hv.z);
  atomicAdd(o + 3, wt * hv.w);
}

// g = leaky_relu(g + bias), float4 vectorized. D4 = D/4.
__global__ __launch_bounds__(256)
void bias_lrelu(float* __restrict__ g, const float* __restrict__ bias,
                int total4, int D4) {
  int i = blockIdx.x * blockDim.x + threadIdx.x;
  if (i >= total4) return;
  float4 v = ((float4*)g)[i];
  float4 b = ((const float4*)bias)[i % D4];
  v.x += b.x; v.y += b.y; v.z += b.z; v.w += b.w;
  v.x = fmaxf(v.x, 0.1f * v.x);
  v.y = fmaxf(v.y, 0.1f * v.y);
  v.z = fmaxf(v.z, 0.1f * v.z);
  v.w = fmaxf(v.w, 0.1f * v.w);
  ((float4*)g)[i] = v;
}

// out = g[N,D] @ Wl[D,C] + bl ; one thread per output element (tiny head).
__global__ __launch_bounds__(256)
void head_gemm(const float* __restrict__ g, const float* __restrict__ Wl,
               const float* __restrict__ bl, float* __restrict__ out,
               int Nn, int D, int C) {
  int idx = blockIdx.x * blockDim.x + threadIdx.x;
  if (idx >= Nn * C) return;
  int row = idx / C, c = idx % C;
  float acc = bl[c];
  const float* gr = g + (size_t)row * D;
  for (int k = 0; k < D; ++k) acc += gr[k] * Wl[k * C + c];
  out[idx] = acc;
}

extern "C" void kernel_launch(void* const* d_in, const int* in_sizes, int n_in,
                              void* d_out, int out_size, void* d_ws, size_t ws_size,
                              hipStream_t stream) {
  const float* x    = (const float*)d_in[0];
  const int*   esrc = (const int*)d_in[1];
  const int*   edst = (const int*)d_in[2];
  const float* ew   = (const float*)d_in[3];
  const float* W1   = (const float*)d_in[4];
  const float* b1   = (const float*)d_in[5];
  const float* W2   = (const float*)d_in[6];
  const float* b2   = (const float*)d_in[7];
  const float* Wl   = (const float*)d_in[8];
  const float* bl   = (const float*)d_in[9];
  float* out = (float*)d_out;

  const int E  = in_sizes[1];
  const int H1 = 512, H2 = 128, C = 7;
  const int F  = in_sizes[4] / H1;       // 1433
  const int Nn = in_sizes[0] / F;        // 30000

  // Workspace layout (reused, peak ~123 MB):
  //   [0, N*H1)       : h1 (GEMM1 out), later h2 (GEMM2 out)
  //   [N*H1, 2*N*H1)  : g1 (SpMM1 out), later g2 (SpMM2 out)
  char* ws = (char*)d_ws;
  float* h1 = (float*)ws;
  float* g1 = (float*)(ws + (size_t)Nn * H1 * sizeof(float));
  float* h2 = h1;
  float* g2 = g1;

  dim3 blk(256);

  // --- Layer 1: h1 = x @ W1 (bf16 WMMA) ---
  gemm_bf16_wmma<<<dim3((Nn + BM - 1) / BM, H1 / BN), blk, 0, stream>>>(
      x, W1, h1, Nn, H1, F);

  // --- SpMM1: g1 = A @ h1 (f32 atomics in L2) ---
  hipMemsetAsync(g1, 0, (size_t)Nn * H1 * sizeof(float), stream);
  {
    long long tot = (long long)E * (H1 / 4);
    spmm_atomic<<<(int)((tot + 255) / 256), blk, 0, stream>>>(
        esrc, edst, ew, h1, g1, E, /*gshift=*/7, H1);
  }
  bias_lrelu<<<((Nn * H1 / 4) + 255) / 256, blk, 0, stream>>>(
      g1, b1, Nn * H1 / 4, H1 / 4);

  // --- Layer 2: h2 = g1 @ W2 (bf16 WMMA) ---
  gemm_bf16_wmma<<<dim3((Nn + BM - 1) / BM, H2 / BN), blk, 0, stream>>>(
      g1, W2, h2, Nn, H2, H1);

  // --- SpMM2: g2 = A @ h2 ---
  hipMemsetAsync(g2, 0, (size_t)Nn * H2 * sizeof(float), stream);
  {
    long long tot = (long long)E * (H2 / 4);
    spmm_atomic<<<(int)((tot + 255) / 256), blk, 0, stream>>>(
        esrc, edst, ew, h2, g2, E, /*gshift=*/5, H2);
  }
  bias_lrelu<<<((Nn * H2 / 4) + 255) / 256, blk, 0, stream>>>(
      g2, b2, Nn * H2 / 4, H2 / 4);

  // --- Head: out = g2 @ Wl + bl ---
  head_gemm<<<((Nn * C) + 255) / 256, blk, 0, stream>>>(
      g2, Wl, bl, out, Nn, H2, C);
}
